// mrun_block_28724741275953
// MI455X (gfx1250) — matrix operand order repaired and checked by reading, not compiled, not run on hardware
//
#include <hip/hip_runtime.h>
#include <hip/hip_bf16.h>

// ---------------------------------------------------------------------------
// Problem constants (from the reference)
// ---------------------------------------------------------------------------
#define Bsz 8
#define Ssz 2048
#define Esz 2048
#define Hsz 8
#define Osz 16
#define HID (4 * Esz)          // 8192
#define CH 128                 // scan chunk length
#define NCHUNK (Ssz / CH)      // 16

typedef __attribute__((ext_vector_type(2)))  float   v2f;
typedef __attribute__((ext_vector_type(8)))  float   v8f;
typedef __attribute__((ext_vector_type(8)))  __bf16  v8bf;
typedef __attribute__((ext_vector_type(16))) __bf16  v16bf;

// D = A(16x4) * B(4x16) + C  (f32 WMMA)
__device__ __forceinline__ v8f wmma4(v2f a, v2f b, v8f c) {
  return __builtin_amdgcn_wmma_f32_16x16x4_f32(false, a, false, b, (short)0, c,
                                               false, false);
}
// D = A(16x32) * B(32x16) + C  (bf16 WMMA, f32 accumulate)
__device__ __forceinline__ v8f wmma32(v16bf a, v16bf b, v8f c) {
  return __builtin_amdgcn_wmma_f32_16x16x32_bf16(false, a, false, b, (short)0,
                                                 c, false, false);
}
// Load a bf16 A-fragment for one lane. Per the ISA 16-bit A layout:
// element j in [0,16): K = (lane<16?0:8) + j + (j>=8?8:0), i.e. two contiguous
// 8-element (16-byte) runs at p and p+16.
__device__ __forceinline__ v16bf ldfragA(const __bf16* p) {
  v8bf lo = *(const v8bf*)(p);
  v8bf hi = *(const v8bf*)(p + 16);
  return __builtin_shufflevector(lo, hi, 0, 1, 2, 3, 4, 5, 6, 7, 8, 9, 10, 11,
                                 12, 13, 14, 15);
}

// ---------------------------------------------------------------------------
// K0: convert f32 weight [Nsize][Ksize] (row-major, B[k][n] = src[n][k]) into
// bf16 B-fragment tiles: dst[((n0/16)*(Ksize/32)+k0/32)*512 + lane*16 + j]
//   = src[n0 + (lane&15)][k0 + (lane<16?0:8) + (j&7) + (j>=8?16:0)]
// so a wave's 16x32 B-fragment is one contiguous 1 KB block (32 B per lane).
// ---------------------------------------------------------------------------
__global__ __launch_bounds__(256) void k_pack(const float* __restrict__ src,
                                              __bf16* __restrict__ dst,
                                              int Nsize, int Ksize) {
  const size_t total = (size_t)Nsize * Ksize;
  const int ktiles = Ksize >> 5;
  for (size_t i = (size_t)blockIdx.x * blockDim.x + threadIdx.x; i < total;
       i += (size_t)gridDim.x * blockDim.x) {
    const size_t tile = i >> 9;
    const int within = (int)(i & 511);
    const int lane = within >> 4;
    const int j = within & 15;
    const int kT = (int)(tile % ktiles);
    const int nT = (int)(tile / ktiles);
    const int n = nT * 16 + (lane & 15);
    const int k = kT * 32 + ((lane < 16) ? 0 : 8) + (j & 7) + ((j >= 8) ? 16 : 0);
    dst[i] = (__bf16)src[(size_t)n * Ksize + k];
  }
}

// ---------------------------------------------------------------------------
// K1: LN1 + up-projection. One 256-thread block per token (b,s).
// new_m[b,s,h][o][p] = sum_c xn[h,o,c] * w_up[p][c] + (o==p)
// stored chain-major: m_ws[((b*H+h)*S + s)*256 + o*16 + p]
// ---------------------------------------------------------------------------
__global__ __launch_bounds__(256) void k_ln1_up(const float* __restrict__ act,
                                                const float* __restrict__ ln1w,
                                                const float* __restrict__ w_up,
                                                float* __restrict__ m_ws) {
  __shared__ float red1[256], red2[256], xn[Esz], wu[256];
  const int tid = threadIdx.x;
  const int token = blockIdx.x;
  const int b = token / Ssz, s = token % Ssz;
  const float* row = act + (size_t)token * Esz;

  float s1 = 0.f, s2 = 0.f;
  for (int e = tid; e < Esz; e += 256) {
    float v = row[e];
    s1 += v;
    s2 += v * v;
  }
  red1[tid] = s1;
  red2[tid] = s2;
  __syncthreads();
  for (int st = 128; st > 0; st >>= 1) {
    if (tid < st) {
      red1[tid] += red1[tid + st];
      red2[tid] += red2[tid + st];
    }
    __syncthreads();
  }
  const float mean = red1[0] * (1.f / Esz);
  const float var = red2[0] * (1.f / Esz) - mean * mean;
  const float rstd = rsqrtf(var + 1e-5f);

  for (int e = tid; e < Esz; e += 256) xn[e] = (row[e] - mean) * rstd * ln1w[e];
  wu[tid] = w_up[tid];  // 16x16 = 256
  __syncthreads();

  for (int idx = tid; idx < Hsz * Osz * Osz; idx += 256) {
    const int h = idx >> 8, o = (idx >> 4) & 15, p = idx & 15;
    float acc = (o == p) ? 1.f : 0.f;
    const float* xr = &xn[h * 256 + o * 16];
    const float* wr = &wu[p * 16];
#pragma unroll
    for (int c = 0; c < 16; ++c) acc += xr[c] * wr[c];
    m_ws[(((size_t)(b * Hsz + h)) * Ssz + s)* 256 + o * 16 + p] = acc;
  }
}

// ---------------------------------------------------------------------------
// K2: chunk-local inclusive scan of 16x16 matrix products. One wave per
// (chain, chunk). Running product kept in C-layout VGPRs; converted to
// A-fragments through LDS each step. Q[s] overwrites M[s] in place.
// ---------------------------------------------------------------------------
__global__ __launch_bounds__(32) void k_local_scan(float* __restrict__ m_ws) {
  __shared__ float mat[256];
  const int lane = threadIdx.x;
  const int r = lane & 15;
  const int h8 = (lane < 16) ? 0 : 8;  // C/D row offset for upper half-wave
  const int h2 = (lane < 16) ? 0 : 2;  // A/B K offset for upper half-wave
  const int chain = blockIdx.x / NCHUNK;
  const int chunk = blockIdx.x % NCHUNK;
  float* base = m_ws + ((size_t)chain * Ssz + (size_t)chunk * CH) * 256;

  v8f run;
#pragma unroll
  for (int i = 0; i < 8; ++i) run[i] = ((i + h8) == r) ? 1.f : 0.f;

  for (int s = 0; s < CH; ++s) {
    float* M = base + (size_t)s * 256;
    v2f bf[4];
#pragma unroll
    for (int k = 0; k < 4; ++k) {  // B[k][n] = M[k][n] (row-major)
      bf[k].x = M[(4 * k + h2) * 16 + r];
      bf[k].y = M[(4 * k + h2 + 1) * 16 + r];
    }
#pragma unroll
    for (int i = 0; i < 8; ++i) mat[(i + h8) * 16 + r] = run[i];
    __syncthreads();
    v2f af[4];
#pragma unroll
    for (int k = 0; k < 4; ++k) {  // A[m][k] from LDS rows
      af[k].x = mat[r * 16 + 4 * k + h2];
      af[k].y = mat[r * 16 + 4 * k + h2 + 1];
    }
    v8f acc;
#pragma unroll
    for (int i = 0; i < 8; ++i) acc[i] = 0.f;
#pragma unroll
    for (int k = 0; k < 4; ++k) acc = wmma4(af[k], bf[k], acc);
    run = acc;
#pragma unroll
    for (int i = 0; i < 8; ++i) M[(i + h8) * 16 + r] = run[i];
    __syncthreads();
  }
}

// ---------------------------------------------------------------------------
// K3: sequential prefix over chunk totals per chain. One wave per chain.
// R_0 = last_state[b,h];  R_{c+1} = R_c @ T_c,  T_c = Q[chunk c, CH-1].
// ---------------------------------------------------------------------------
__global__ __launch_bounds__(32) void k_chunk_scan(
    const float* __restrict__ m_ws, const float* __restrict__ last_state,
    float* __restrict__ r_ws) {
  __shared__ float mat[256];
  const int lane = threadIdx.x;
  const int r = lane & 15;
  const int h8 = (lane < 16) ? 0 : 8;
  const int h2 = (lane < 16) ? 0 : 2;
  const int chain = blockIdx.x;
  const float* L = last_state + (size_t)chain * 256;

  v8f run;
#pragma unroll
  for (int i = 0; i < 8; ++i) run[i] = L[(i + h8) * 16 + r];

  for (int c = 0; c < NCHUNK; ++c) {
    float* R = r_ws + ((size_t)chain * NCHUNK + c) * 256;
#pragma unroll
    for (int i = 0; i < 8; ++i) R[(i + h8) * 16 + r] = run[i];
    if (c == NCHUNK - 1) break;
    const float* T =
        m_ws + ((size_t)chain * Ssz + (size_t)c * CH + (CH - 1)) * 256;
    v2f bf[4];
#pragma unroll
    for (int k = 0; k < 4; ++k) {
      bf[k].x = T[(4 * k + h2) * 16 + r];
      bf[k].y = T[(4 * k + h2 + 1) * 16 + r];
    }
#pragma unroll
    for (int i = 0; i < 8; ++i) mat[(i + h8) * 16 + r] = run[i];
    __syncthreads();
    v2f af[4];
#pragma unroll
    for (int k = 0; k < 4; ++k) {
      af[k].x = mat[r * 16 + 4 * k + h2];
      af[k].y = mat[r * 16 + 4 * k + h2 + 1];
    }
    v8f acc;
#pragma unroll
    for (int i = 0; i < 8; ++i) acc[i] = 0.f;
#pragma unroll
    for (int k = 0; k < 4; ++k) acc = wmma4(af[k], bf[k], acc);
    run = acc;
    __syncthreads();
  }
}

// ---------------------------------------------------------------------------
// K4: finalize scan + down-projection + residual. One wave per (chain,chunk).
// states = R_chunk @ Q[s];  mru = states @ w_down^T;
// out = activations + mru ; for b==B-1 also write states to new_state tail.
// ---------------------------------------------------------------------------
__global__ __launch_bounds__(32) void k_finalize(
    const float* __restrict__ m_ws, const float* __restrict__ r_ws,
    const float* __restrict__ act, const float* __restrict__ w_down,
    float* __restrict__ out) {
  __shared__ float mat[256];
  const int lane = threadIdx.x;
  const int r = lane & 15;
  const int h8 = (lane < 16) ? 0 : 8;
  const int h2 = (lane < 16) ? 0 : 2;
  const int chain = blockIdx.x / NCHUNK;
  const int chunk = blockIdx.x % NCHUNK;
  const int b = chain / Hsz, h = chain % Hsz;

  const float* R = r_ws + ((size_t)chain * NCHUNK + chunk) * 256;
  v2f aR[4];
#pragma unroll
  for (int k = 0; k < 4; ++k) {  // A[m=o][k=p] = R[o][p]
    aR[k].x = R[r * 16 + 4 * k + h2];
    aR[k].y = R[r * 16 + 4 * k + h2 + 1];
  }
  v2f bw[4];
#pragma unroll
  for (int k = 0; k < 4; ++k) {  // B[k=p][n=c] = w_down[c][p]
    bw[k].x = w_down[r * 16 + 4 * k + h2];
    bw[k].y = w_down[r * 16 + 4 * k + h2 + 1];
  }

  for (int sl = 0; sl < CH; ++sl) {
    const int sg = chunk * CH + sl;
    const float* Q = m_ws + ((size_t)chain * Ssz + sg) * 256;
    v2f bq[4];
#pragma unroll
    for (int k = 0; k < 4; ++k) {
      bq[k].x = Q[(4 * k + h2) * 16 + r];
      bq[k].y = Q[(4 * k + h2 + 1) * 16 + r];
    }
    v8f st;
#pragma unroll
    for (int i = 0; i < 8; ++i) st[i] = 0.f;
#pragma unroll
    for (int k = 0; k < 4; ++k) st = wmma4(aR[k], bq[k], st);

    if (b == Bsz - 1) {  // new_state = states[B-1] : (S,H,O,O) tail of d_out
      float* ns =
          out + (size_t)Bsz * Ssz * Esz + ((size_t)sg * Hsz + h) * 256;
#pragma unroll
      for (int i = 0; i < 8; ++i) ns[(i + h8) * 16 + r] = st[i];
    }
#pragma unroll
    for (int i = 0; i < 8; ++i) mat[(i + h8) * 16 + r] = st[i];
    __syncthreads();
    v2f a2[4];
#pragma unroll
    for (int k = 0; k < 4; ++k) {
      a2[k].x = mat[r * 16 + 4 * k + h2];
      a2[k].y = mat[r * 16 + 4 * k + h2 + 1];
    }
    v8f mr;
#pragma unroll
    for (int i = 0; i < 8; ++i) mr[i] = 0.f;
#pragma unroll
    for (int k = 0; k < 4; ++k) mr = wmma4(a2[k], bw[k], mr);

    const size_t obase = ((size_t)b * Ssz + sg) * Esz + (size_t)h * 256;
#pragma unroll
    for (int i = 0; i < 8; ++i) {
      const int o = i + h8;
      out[obase + o * 16 + r] = act[obase + o * 16 + r] + mr[i];
    }
    __syncthreads();
  }
}

// ---------------------------------------------------------------------------
// K5: LN2 + fused MLP (bf16 WMMA, f32 accumulate), residual into d_out.
// 256 threads (8 waves) per 16-token tile. Dynamic LDS:
//   red1/red2/mv | xn bf16 16 x 2064 | g bf16 16 x 1040
// Each wave: GEMM1 -> computes 8 hidden 16x16 tiles per 1024-f chunk
// (64 bf16 WMMA each), exact-erf GELU, stage in LDS; GEMM2 -> accumulates its
// private 16 tokens x 256 output columns (16 v8f accumulators in VGPRs).
// ---------------------------------------------------------------------------
#define XSTR 2064  // 2048 + 16 bf16 pad (32 B) to stagger LDS banks
#define GSTR 1040  // 1024 + 16 bf16 pad
#define FCH 1024   // hidden-dim chunk staged in LDS

__global__ __launch_bounds__(256) void k_ln2_mlp(
    float* __restrict__ out, const float* __restrict__ ln2w,
    const __bf16* __restrict__ w1p, const __bf16* __restrict__ w2p) {
  extern __shared__ char smem[];
  float* red1 = (float*)smem;          // 16*17
  float* red2 = red1 + 16 * 17;        // 16*17
  float* mv = red2 + 16 * 17;          // 16*2
  __bf16* xn = (__bf16*)(mv + 32);     // 16*XSTR
  __bf16* gl = xn + 16 * XSTR;         // 16*GSTR

  const int tid = threadIdx.x;
  const int t0 = blockIdx.x * 16;
  float* act = out + (size_t)t0 * Esz;

  // LN2 statistics: 16 partial lanes per token
  {
    const int tk = tid >> 4, part = tid & 15;
    const float* rp = act + (size_t)tk * Esz;
    float s1 = 0.f, s2 = 0.f;
    for (int e = part; e < Esz; e += 16) {
      float v = rp[e];
      s1 += v;
      s2 += v * v;
    }
    red1[tk * 17 + part] = s1;
    red2[tk * 17 + part] = s2;
  }
  __syncthreads();
  if (tid < 16) {
    float s1 = 0.f, s2 = 0.f;
    for (int p = 0; p < 16; ++p) {
      s1 += red1[tid * 17 + p];
      s2 += red2[tid * 17 + p];
    }
    const float mean = s1 * (1.f / Esz);
    const float var = s2 * (1.f / Esz) - mean * mean;
    mv[tid * 2] = mean;
    mv[tid * 2 + 1] = rsqrtf(var + 1e-5f);
  }
  __syncthreads();
  for (int idx = tid; idx < 16 * Esz; idx += 256) {
    const int tk = idx >> 11, e = idx & (Esz - 1);
    xn[tk * XSTR + e] = (__bf16)((act[(size_t)tk * Esz + e] - mv[tk * 2]) *
                                 mv[tk * 2 + 1] * ln2w[e]);
  }
  __syncthreads();

  const int wave = tid >> 5, lane = tid & 31;
  const int r = lane & 15;
  const int hk = (lane < 16) ? 0 : 8;  // A-frag K offset AND C-row offset

  v8f accO[16];
#pragma unroll
  for (int t = 0; t < 16; ++t)
#pragma unroll
    for (int i = 0; i < 8; ++i) accO[t][i] = 0.f;

  for (int fc0 = 0; fc0 < HID; fc0 += FCH) {
    // ---- GEMM1 + GELU: wave computes f columns [fc0+wave*128, +128) ----
#pragma unroll 1
    for (int ft = 0; ft < 8; ++ft) {
      const int f0 = fc0 + wave * 128 + ft * 16;
      v8f acc;
#pragma unroll
      for (int i = 0; i < 8; ++i) acc[i] = 0.f;
      const v16bf* wp =
          (const v16bf*)w1p + ((size_t)(f0 >> 4) * (Esz / 32)) * 32 + lane;
      const __bf16* xp = xn + r * XSTR + hk;
#pragma unroll 4
      for (int e0 = 0; e0 < Esz; e0 += 32) {
        v16bf a = ldfragA(xp + e0);
        v16bf bb = wp[(e0 >> 5) * 32];
        acc = wmma32(a, bb, acc);
      }
#pragma unroll
      for (int i = 0; i < 8; ++i) {
        const float x = acc[i];
        const float g = 0.5f * x * (1.f + erff(x * 0.70710678f));
        gl[(i + hk) * GSTR + (wave * 128 + ft * 16 + r)] = (__bf16)g;
      }
    }
    __syncthreads();
    // ---- GEMM2: wave accumulates its 256 output columns over this chunk ----
#pragma unroll
    for (int t = 0; t < 16; ++t) {
      const int eo = wave * 256 + t * 16;
      const v16bf* wp2 =
          (const v16bf*)w2p + ((size_t)(eo >> 4) * (HID / 32)) * 32 + lane;
      const __bf16* gp = gl + r * GSTR + hk;
      v8f acc = accO[t];
#pragma unroll 4
      for (int k = 0; k < FCH / 32; ++k) {
        v16bf a = ldfragA(gp + k * 32);
        v16bf bb = wp2[((fc0 + k * 32) >> 5) * 32];
        acc = wmma32(a, bb, acc);
      }
      accO[t] = acc;
    }
    __syncthreads();
  }

  // ---- residual epilogue: out += mlp ----
#pragma unroll
  for (int t = 0; t < 16; ++t) {
    const int eo = wave * 256 + t * 16;
#pragma unroll
    for (int i = 0; i < 8; ++i) {
      const size_t o = (size_t)(t0 + i + hk) * Esz + eo + r;
      out[o] = out[o] + accO[t][i];
    }
  }
}

// ---------------------------------------------------------------------------
// Launch
// ---------------------------------------------------------------------------
extern "C" void kernel_launch(void* const* d_in, const int* in_sizes, int n_in,
                              void* d_out, int out_size, void* d_ws,
                              size_t ws_size, hipStream_t stream) {
  const float* activ = (const float*)d_in[0];       // (B,S,E)
  const float* last_state = (const float*)d_in[1];  // (B,H,O,O)
  const float* ln1w = (const float*)d_in[2];        // (E)
  const float* ln2w = (const float*)d_in[3];        // (E)
  const float* w_up = (const float*)d_in[4];        // (O,C)
  const float* w_down = (const float*)d_in[5];      // (C,O)
  const float* mlp_w1 = (const float*)d_in[6];      // (4E,E)
  const float* mlp_w2 = (const float*)d_in[7];      // (E,4E)
  float* out = (float*)d_out;

  // workspace layout
  const size_t M_ELEMS = (size_t)Bsz * Ssz * Hsz * 256;      // new_m / Q
  const size_t R_ELEMS = (size_t)Bsz * Hsz * NCHUNK * 256;   // chunk prefixes
  float* m_ws = (float*)d_ws;
  float* r_ws = m_ws + M_ELEMS;
  __bf16* w1p = (__bf16*)(r_ws + R_ELEMS);
  __bf16* w2p = w1p + (size_t)HID * Esz;

  // pre-pack MLP weights into bf16 B-fragment tiles
  k_pack<<<2048, 256, 0, stream>>>(mlp_w1, w1p, HID, Esz);
  k_pack<<<2048, 256, 0, stream>>>(mlp_w2, w2p, Esz, HID);

  // LN1 + up-projection
  k_ln1_up<<<Bsz * Ssz, 256, 0, stream>>>(activ, ln1w, w_up, m_ws);

  // blocked associative scan of 16x16 matrix products
  k_local_scan<<<Bsz * Hsz * NCHUNK, 32, 0, stream>>>(m_ws);
  k_chunk_scan<<<Bsz * Hsz, 32, 0, stream>>>(m_ws, last_state, r_ws);
  k_finalize<<<Bsz * Hsz * NCHUNK, 32, 0, stream>>>(m_ws, r_ws, activ, w_down,
                                                    out);

  // LN2 + fused MLP + residual
  const size_t smem_bytes =
      (size_t)(16 * 17 * 2 + 32) * sizeof(float) +
      (size_t)(16 * XSTR + 16 * GSTR) * sizeof(__bf16);  // ~101.6 KB
  (void)hipFuncSetAttribute((const void*)k_ln2_mlp,
                            hipFuncAttributeMaxDynamicSharedMemorySize,
                            (int)smem_bytes);
  k_ln2_mlp<<<(Bsz * Ssz) / 16, 256, smem_bytes, stream>>>(out, ln2w, w1p,
                                                           w2p);
}
